// tFocusedLinearAttention_60782377173174
// MI455X (gfx1250) — matrix operand order repaired and checked by reading, not compiled, not run on hardware
//
#include <hip/hip_runtime.h>
#include <hip/hip_bf16.h>

// ---------------------------------------------------------------------------
// Focused Linear Attention forward, MI455X (gfx1250, wave32, WMMA).
// GEMMs (q, k, out-proj) run as f16 WMMA (v_wmma_f32_16x16x32_f16) with f32
// accumulation; everything else stays f32. Memory-bound overall (~1.3 GB of
// HBM traffic at 23.3 TB/s). Weight tiles stream into a double-buffered LDS
// tile via GLOBAL_LOAD_ASYNC_TO_LDS_B128 (ASYNCcnt), one barrier per K-step;
// A fragments are software-pipelined one K-step ahead in VGPRs.
// ---------------------------------------------------------------------------

typedef _Float16 half_t;
typedef __attribute__((ext_vector_type(16))) _Float16 v16h;
typedef __attribute__((ext_vector_type(8)))  _Float16 v8h;
typedef __attribute__((ext_vector_type(8)))  float    v8f;
typedef int v4i_ __attribute__((vector_size(16)));   // matches builtin param type

#define BB 8
#define NN 4096
#define CC 768
#define MM (BB * NN)
#define NHEADS 8
#define HDIM 96
#define SCALE_FACTOR 0.10206207261596575f   // 96^-0.5

#if defined(__has_builtin)
# if __has_builtin(__builtin_amdgcn_global_load_async_to_lds_b128)
#  define HAS_ASYNC_COPY 1
# endif
#endif
#ifndef HAS_ASYNC_COPY
# define HAS_ASYNC_COPY 0
#endif

static __device__ __forceinline__ v16h cat8(v8h a, v8h b) {
    return __builtin_shufflevector(a, b, 0, 1, 2, 3, 4, 5, 6, 7,
                                          8, 9, 10, 11, 12, 13, 14, 15);
}

#if HAS_ASYNC_COPY
// Issue the 32B this thread owns of the 128x32 weight tile as two async
// b128 copies (per-wave: 2 outstanding ASYNCcnt entries per tile).
typedef __attribute__((address_space(1))) v4i_ gvec_t;
typedef __attribute__((address_space(3))) v4i_ lvec_t;

static __device__ __forceinline__ void stage_async(const half_t* g, half_t* l) {
    gvec_t* gp = (gvec_t*)g;
    lvec_t* lp = (lvec_t*)l;
    __builtin_amdgcn_global_load_async_to_lds_b128(gp, lp, 0, 0);
    __builtin_amdgcn_global_load_async_to_lds_b128(gp, lp, 16, 0);
}

static __device__ __forceinline__ void wait_async0() {
#if __has_builtin(__builtin_amdgcn_s_wait_asynccnt)
    __builtin_amdgcn_s_wait_asynccnt(0);
#else
    asm volatile("s_wait_asynccnt 0x0" ::: "memory");
#endif
}
#endif

// ---------------------------------------------------------------------------
// Kernel: f32 -> f16 convert
// ---------------------------------------------------------------------------
__global__ void f32_to_f16_k(const float* __restrict__ in,
                             half_t* __restrict__ out, int n) {
    int i = blockIdx.x * blockDim.x + threadIdx.x;
    if (i < n) out[i] = (half_t)in[i];
}

// ---------------------------------------------------------------------------
// Kernel: softplus(scale_p)
// ---------------------------------------------------------------------------
__global__ void softplus_k(const float* __restrict__ p,
                           float* __restrict__ s, int n) {
    int i = blockIdx.x * blockDim.x + threadIdx.x;
    if (i < n) {
        float v = p[i];
        s[i] = fmaxf(v, 0.0f) + log1pf(expf(-fabsf(v)));
    }
}

// ---------------------------------------------------------------------------
// WMMA GEMM:  out[M,N] = A[M,K](f16) * W[N,K](f16)^T  (+ bias[N] if non-null)
// Block = 256 threads = 8 waves. Each wave: 16 rows x 128 cols.
// Weight tile (32K x 128N) in LDS, padded row stride 40 halves (80 B,
// 16B-aligned b128 accesses, conflict-free 20-dword bank step).
// Async pipeline per K-step: wait(own tile resident) -> barrier(all waves) ->
// issue async stage of next tile into other buffer -> 8x WMMA. A wave past
// the barrier has finished the previous iteration's reads, so one barrier per
// step suffices. A fragments are prefetched one K-step ahead into VGPRs.
// ---------------------------------------------------------------------------
#define LDSB_STRIDE 40
#define LDSB_TILE   (128 * LDSB_STRIDE)

__global__ __launch_bounds__(256)
void gemm_f16f32_k(const half_t* __restrict__ A, const half_t* __restrict__ W,
                   const float* __restrict__ bias, float* __restrict__ out,
                   int M, int N, int K) {
#if HAS_ASYNC_COPY
    __shared__ half_t ldsB[2 * LDSB_TILE];
#else
    __shared__ half_t ldsB[LDSB_TILE];
#endif

    const int wave = threadIdx.x >> 5;
    const int lane = threadIdx.x & 31;
    const int rowBase = blockIdx.y * 128 + wave * 16;
    const int colBase = blockIdx.x * 128;

    // A fragment mapping (16-bit A 16x32): lanes 0-15 hold K 0..7 & 16..23,
    // lanes 16-31 hold K 8..15 & 24..31; row = lane&15.
    const int arow  = rowBase + (lane & 15);
    const int akoff = 8 * (lane >> 4);
    const half_t* __restrict__ aptr = A + (size_t)arow * K;

    // B fragment mapping: col = lane&15 within tile; K half by lane>>4.
    const int bn = lane & 15;
    const int bk = 8 * (lane >> 4);

    v8f acc[8];
    const v8f vzero = {0.f, 0.f, 0.f, 0.f, 0.f, 0.f, 0.f, 0.f};
#pragma unroll
    for (int t = 0; t < 8; ++t) acc[t] = vzero;

    const int ldn = threadIdx.x >> 1;        // 0..127 : column within tile
    const int ldh = threadIdx.x & 1;         // which 16-half of K
    const half_t* __restrict__ wrow = W + (size_t)(colBase + ldn) * K + 16 * ldh;
    half_t* __restrict__ ldst0 = &ldsB[ldn * LDSB_STRIDE + 16 * ldh];

#if HAS_ASYNC_COPY
    const int nk = K >> 5;
    // Prologue: tile 0 -> buffer 0 (async), A fragment 0 -> VGPRs.
    stage_async(wrow, ldst0);
    v8h alo = *(const v8h*)(aptr + akoff);
    v8h ahi = *(const v8h*)(aptr + 16 + akoff);

    for (int it = 0; it < nk; ++it) {
        const int kk = it << 5;
        const half_t* __restrict__ curbuf = &ldsB[(it & 1) * LDSB_TILE];

        wait_async0();     // own async copies of tile it have landed in LDS
        __syncthreads();   // all waves' copies landed; all waves done reading
                           // the other buffer (they passed last iteration)

        if (it + 1 < nk) {
            // Stage tile it+1 into the other buffer; overlaps the WMMAs below.
            stage_async(wrow + kk + 32, ldst0 + (((it + 1) & 1) ? LDSB_TILE : 0));
        }

        v16h afrag = cat8(alo, ahi);
        if (it + 1 < nk) {
            // Pipeline the A stream: issue next fragment's loads now, waited
            // on only at the next iteration's cat8 use.
            alo = *(const v8h*)(aptr + kk + 32 + akoff);
            ahi = *(const v8h*)(aptr + kk + 48 + akoff);
            __builtin_prefetch(aptr + kk + 64, 0, 3);   // near-scope, 2 ahead
        }

#pragma unroll
        for (int t = 0; t < 8; ++t) {
            const half_t* bp = &curbuf[(t * 16 + bn) * LDSB_STRIDE + bk];
            v8h b0 = *(const v8h*)bp;
            v8h b1 = *(const v8h*)(bp + 16);
            v16h bfrag = cat8(b0, b1);
            acc[t] = __builtin_amdgcn_wmma_f32_16x16x32_f16(
                false, afrag, false, bfrag, (short)0, acc[t], false, false);
        }
    }
#else
    for (int kk = 0; kk < K; kk += 32) {
        __syncthreads();
        {
            v8h w0 = *(const v8h*)(wrow + kk);
            v8h w1 = *(const v8h*)(wrow + kk + 8);
            *(v8h*)(ldst0)     = w0;
            *(v8h*)(ldst0 + 8) = w1;
        }
        __syncthreads();

        v8h alo = *(const v8h*)(aptr + kk + akoff);
        v8h ahi = *(const v8h*)(aptr + kk + 16 + akoff);
        v16h afrag = cat8(alo, ahi);

#pragma unroll
        for (int t = 0; t < 8; ++t) {
            const half_t* bp = &ldsB[(t * 16 + bn) * LDSB_STRIDE + bk];
            v8h b0 = *(const v8h*)bp;
            v8h b1 = *(const v8h*)(bp + 16);
            v16h bfrag = cat8(b0, b1);
            acc[t] = __builtin_amdgcn_wmma_f32_16x16x32_f16(
                false, afrag, false, bfrag, (short)0, acc[t], false, false);
        }
    }
#endif

    // Epilogue: C/D layout — VGPR i: lanes 0-15 -> M=i, lanes 16-31 -> M=8+i;
    // N = lane&15.
    const int mbase = 8 * (lane >> 4);
    const int nlane = lane & 15;
#pragma unroll
    for (int t = 0; t < 8; ++t) {
        int col = colBase + t * 16 + nlane;
        float bv = bias ? bias[col] : 0.0f;
#pragma unroll
        for (int i = 0; i < 8; ++i) {
            int r = rowBase + mbase + i;
            out[(size_t)r * N + col] = acc[t][i] + bv;
        }
    }
}

// ---------------------------------------------------------------------------
// Kernel: A_raw[row] = (q[row,:] . w_g) * d^-0.5     (one wave per row)
// ---------------------------------------------------------------------------
__global__ __launch_bounds__(256)
void araw_k(const float* __restrict__ q, const float* __restrict__ wg,
            float* __restrict__ araw, int rows) {
    int wave = threadIdx.x >> 5;
    int lane = threadIdx.x & 31;
    int row = blockIdx.x * 8 + wave;
    if (row >= rows) return;
    const float* qr = q + (size_t)row * CC;
    float s = 0.0f;
    for (int c = lane; c < CC; c += 32) s += qr[c] * wg[c];
#pragma unroll
    for (int off = 16; off > 0; off >>= 1) s += __shfl_xor(s, off, 32);
    if (lane == 0) araw[row] = s * SCALE_FACTOR;
}

// ---------------------------------------------------------------------------
// Kernel: inv[b] = 1 / max(||A[b,:]||_2, 1e-12)   (one block per batch)
// ---------------------------------------------------------------------------
__global__ __launch_bounds__(256)
void anorm_k(const float* __restrict__ araw, float* __restrict__ inv) {
    __shared__ float red[256];
    int b = blockIdx.x;
    float s = 0.0f;
    for (int n = threadIdx.x; n < NN; n += 256) {
        float a = araw[b * NN + n];
        s += a * a;
    }
    red[threadIdx.x] = s;
    __syncthreads();
    for (int o = 128; o > 0; o >>= 1) {
        if (threadIdx.x < o) red[threadIdx.x] += red[threadIdx.x + o];
        __syncthreads();
    }
    if (threadIdx.x == 0) inv[b] = 1.0f / fmaxf(sqrtf(red[0]), 1e-12f);
}

// ---------------------------------------------------------------------------
// Kernel: focusing feature map per row (b,n):
//   qf = (relu(q)+1e-6)/scale ; kf = (relu(k0+pos)+1e-6)/scale
//   q'' = qf^3 * sqrt(sum qf^2 / sum qf^6)   (same for k)
// One 256-thread block per row, 3 channels per thread.
// ---------------------------------------------------------------------------
__global__ __launch_bounds__(256)
void featmap_k(const float* __restrict__ q, const float* __restrict__ k0,
               const float* __restrict__ pos, const float* __restrict__ scale,
               float* __restrict__ q2, float* __restrict__ k2) {
    __shared__ float red[4][256];
    int row = blockIdx.x;
    int n = row & (NN - 1);
    int tid = threadIdx.x;
    const float* qr = q  + (size_t)row * CC;
    const float* kr = k0 + (size_t)row * CC;
    const float* pr = pos + (size_t)n * CC;

    float qf[3], kf[3];
    float sq2 = 0.f, sq6 = 0.f, sk2 = 0.f, sk6 = 0.f;
#pragma unroll
    for (int j = 0; j < 3; ++j) {
        int c = tid + j * 256;
        float sc = scale[c];
        float qv = (fmaxf(qr[c], 0.0f) + 1e-6f) / sc;
        float kv = (fmaxf(kr[c] + pr[c], 0.0f) + 1e-6f) / sc;
        qf[j] = qv; kf[j] = kv;
        float q3 = qv * qv * qv, k3 = kv * kv * kv;
        sq2 += qv * qv; sq6 += q3 * q3;
        sk2 += kv * kv; sk6 += k3 * k3;
    }
    red[0][tid] = sq2; red[1][tid] = sq6; red[2][tid] = sk2; red[3][tid] = sk6;
    __syncthreads();
    for (int o = 128; o > 0; o >>= 1) {
        if (tid < o) {
            red[0][tid] += red[0][tid + o];
            red[1][tid] += red[1][tid + o];
            red[2][tid] += red[2][tid + o];
            red[3][tid] += red[3][tid + o];
        }
        __syncthreads();
    }
    float rq = sqrtf(red[0][0] / red[1][0]);
    float rk = sqrtf(red[2][0] / red[3][0]);
    float* q2r = q2 + (size_t)row * CC;
    float* k2r = k2 + (size_t)row * CC;
#pragma unroll
    for (int j = 0; j < 3; ++j) {
        int c = tid + j * 256;
        q2r[c] = qf[j] * qf[j] * qf[j] * rq;
        k2r[c] = kf[j] * kf[j] * kf[j] * rk;
    }
}

// ---------------------------------------------------------------------------
// Chunked n-reduction (no atomics -> deterministic):
//   PG[z,b,c] = sum_{n in chunk z} A_raw[b,n] * q[b,n,c]
//   PK[z,b,c] = sum_{n in chunk z} k''[b,n,c]
// ---------------------------------------------------------------------------
#define NCHUNK 16
__global__ __launch_bounds__(256)
void reduce_partial_k(const float* __restrict__ q, const float* __restrict__ k2,
                      const float* __restrict__ araw,
                      float* __restrict__ PG, float* __restrict__ PK) {
    int b = blockIdx.y;
    int z = blockIdx.z;
    int c = blockIdx.x * 256 + threadIdx.x;
    const int per = NN / NCHUNK;
    int n0 = z * per;
    const float* qb = q  + ((size_t)b * NN + n0) * CC + c;
    const float* kb = k2 + ((size_t)b * NN + n0) * CC + c;
    const float* ab = araw + b * NN + n0;
    float g = 0.f, km = 0.f;
    for (int n = 0; n < per; ++n) {
        g  += ab[n] * qb[(size_t)n * CC];
        km += kb[(size_t)n * CC];
    }
    PG[((size_t)z * BB + b) * CC + c] = g;
    PK[((size_t)z * BB + b) * CC + c] = km;
}

__global__ __launch_bounds__(256)
void reduce_final_k(const float* __restrict__ PG, const float* __restrict__ PK,
                    const float* __restrict__ inv,
                    float* __restrict__ G, float* __restrict__ kmean) {
    int b = blockIdx.y;
    int c = blockIdx.x * 256 + threadIdx.x;
    float g = 0.f, km = 0.f;
#pragma unroll
    for (int z = 0; z < NCHUNK; ++z) {
        g  += PG[((size_t)z * BB + b) * CC + c];
        km += PK[((size_t)z * BB + b) * CC + c];
    }
    G[b * CC + c] = g * inv[b];
    kmean[b * CC + c] = km * (1.0f / (float)NN);
}

// ---------------------------------------------------------------------------
// Kernel: per-head z = 1/(q''.k_mean + 1e-6); out_pre = G*k0*z  -> f16
// One 768-thread block per row (one thread per channel).
// ---------------------------------------------------------------------------
__global__ __launch_bounds__(768)
void zxg_k(const float* __restrict__ q2, const float* __restrict__ k0,
           const float* __restrict__ kmean, const float* __restrict__ G,
           half_t* __restrict__ outp) {
    __shared__ float p[CC];
    __shared__ float zz[NHEADS];
    int row = blockIdx.x;
    int b = row / NN;
    int c = threadIdx.x;
    p[c] = q2[(size_t)row * CC + c] * kmean[b * CC + c];
    __syncthreads();
    if (c < NHEADS) {
        float s = 0.f;
        for (int d = 0; d < HDIM; ++d) s += p[c * HDIM + d];
        zz[c] = 1.0f / (s + 1e-6f);
    }
    __syncthreads();
    float val = G[b * CC + c] * k0[(size_t)row * CC + c] * zz[c / HDIM];
    outp[(size_t)row * CC + c] = (half_t)val;
}

// ---------------------------------------------------------------------------
// Host-side launch
// ---------------------------------------------------------------------------
extern "C" void kernel_launch(void* const* d_in, const int* in_sizes, int n_in,
                              void* d_out, int out_size, void* d_ws, size_t ws_size,
                              hipStream_t stream) {
    const float* x       = (const float*)d_in[0];
    const float* Wq      = (const float*)d_in[3];
    const float* Wkv     = (const float*)d_in[4];
    const float* Wproj   = (const float*)d_in[5];
    const float* bproj   = (const float*)d_in[6];
    const float* w_g     = (const float*)d_in[7];
    const float* scale_p = (const float*)d_in[8];
    const float* pos_enc = (const float*)d_in[9];
    float* out = (float*)d_out;

    char* w = (char*)d_ws;
    size_t o = 0;
    auto take = [&](size_t bytes) {
        void* p = w + o;
        o += (bytes + 255) & ~(size_t)255;
        return p;
    };
    half_t* X16   = (half_t*)take((size_t)MM * CC * 2);
    half_t* WQ16  = (half_t*)take((size_t)CC * CC * 2);
    half_t* WKV16 = (half_t*)take((size_t)CC * CC * 2);
    half_t* WP16  = (half_t*)take((size_t)CC * CC * 2);
    float*  SCALE = (float*)take((size_t)CC * 4);
    float*  ARAW  = (float*)take((size_t)MM * 4);
    float*  INV   = (float*)take((size_t)BB * 4);
    float*  Gbuf  = (float*)take((size_t)BB * CC * 4);
    float*  KMEAN = (float*)take((size_t)BB * CC * 4);
    float*  PG    = (float*)take((size_t)NCHUNK * BB * CC * 4);
    float*  PK    = (float*)take((size_t)NCHUNK * BB * CC * 4);
    float*  Q     = (float*)take((size_t)MM * CC * 4);
    float*  K0    = (float*)take((size_t)MM * CC * 4);
    float*  Q2    = (float*)take((size_t)MM * CC * 4);
    float*  K2    = (float*)take((size_t)MM * CC * 4);
    half_t* OUTP16 = (half_t*)K2;   // K2 is dead after reduce_partial_k

    // 1) precision conversion
    {
        int n = MM * CC;
        f32_to_f16_k<<<(n + 255) / 256, 256, 0, stream>>>(x, X16, n);
        int nw = CC * CC;
        f32_to_f16_k<<<(nw + 255) / 256, 256, 0, stream>>>(Wq, WQ16, nw);
        f32_to_f16_k<<<(nw + 255) / 256, 256, 0, stream>>>(Wkv, WKV16, nw);
        f32_to_f16_k<<<(nw + 255) / 256, 256, 0, stream>>>(Wproj, WP16, nw);
        softplus_k<<<(CC + 255) / 256, 256, 0, stream>>>(scale_p, SCALE, CC);
    }

    dim3 gg(CC / 128, MM / 128);   // (6, 256)

    // 2) q = x Wq^T ; k0 = x Wkv^T   (f16 WMMA, f32 out)
    gemm_f16f32_k<<<gg, 256, 0, stream>>>(X16, WQ16,  nullptr, Q,  MM, CC, CC);
    gemm_f16f32_k<<<gg, 256, 0, stream>>>(X16, WKV16, nullptr, K0, MM, CC, CC);

    // 3) aggregation weights
    araw_k<<<MM / 8, 256, 0, stream>>>(Q, w_g, ARAW, MM);
    anorm_k<<<BB, 256, 0, stream>>>(ARAW, INV);

    // 4) focusing feature map
    featmap_k<<<MM, 256, 0, stream>>>(Q, K0, pos_enc, SCALE, Q2, K2);

    // 5) G and k_mean (chunked deterministic reduction over n)
    reduce_partial_k<<<dim3(CC / 256, BB, NCHUNK), 256, 0, stream>>>(Q, K2, ARAW, PG, PK);
    reduce_final_k<<<dim3(CC / 256, BB), 256, 0, stream>>>(PG, PK, INV, Gbuf, KMEAN);

    // 6) z + gated value, write f16 for the projection GEMM
    zxg_k<<<MM, 768, 0, stream>>>(Q2, K0, KMEAN, Gbuf, OUTP16);

    // 7) out = out_pre Wproj^T + bproj
    gemm_f16f32_k<<<gg, 256, 0, stream>>>(OUTP16, WP16, bproj, out, MM, CC, CC);

    (void)in_sizes; (void)n_in; (void)out_size; (void)ws_size;
}
